// SelfAttention_13864154431841
// MI455X (gfx1250) — matrix operand order repaired and checked by reading, not compiled, not run on hardware
//
#include <hip/hip_runtime.h>

typedef __attribute__((ext_vector_type(16))) _Float16 v16h;
typedef __attribute__((ext_vector_type(8)))  _Float16 v8h;
typedef __attribute__((ext_vector_type(8)))  float    v8f;

#define HEADS 8
#define DIMH  64
#define SEQ   4096
#define DIM   512
#define SCALE 0.125f   // 64^-0.5
#define QT    2        // query tiles per wave (32 queries)
#define KT64  (SEQ / 64)
#define LPAD  72       // LDS row pitch in halves (64 + 8 pad, keeps 16B align)

union F16x16 {
  v16h v;
  v8h h2[2];
  _Float16 e[16];
  unsigned u[8];       // u[j] = halves e[2j] (lo) , e[2j+1] (hi)
};

static __device__ __forceinline__ v8f zero8() {
  v8f z;
#pragma unroll
  for (int i = 0; i < 8; ++i) z[i] = 0.0f;
  return z;
}

static __device__ __forceinline__ v16h ld2x8(const _Float16* p0, const _Float16* p1) {
  F16x16 u;
  u.h2[0] = *reinterpret_cast<const v8h*>(p0);
  u.h2[1] = *reinterpret_cast<const v8h*>(p1);
  return u.v;
}
static __device__ __forceinline__ v16h ld16(const _Float16* p) { return ld2x8(p, p + 8); }

static __device__ __forceinline__ v16h cvt2x8(const float* p0, const float* p1) {
  F16x16 u;
#pragma unroll
  for (int i = 0; i < 8; ++i) { u.e[i] = (_Float16)p0[i]; u.e[8 + i] = (_Float16)p1[i]; }
  return u.v;
}
static __device__ __forceinline__ v16h cvt16(const float* p) { return cvt2x8(p, p + 8); }

// pack two f32 into one u32 of f16 halves (lowers to v_cvt_pk_f16_f32)
static __device__ __forceinline__ unsigned pack2(float a, float b) {
  union { unsigned u; _Float16 h[2]; } x;
  x.h[0] = (_Float16)a;
  x.h[1] = (_Float16)b;
  return x.u;
}

static __device__ __forceinline__ v8f wmma16(v16h a, v16h b, v8f c) {
  // D(f32 16x16) = A(16x32 f16) x B(32x16 f16) + C
  return __builtin_amdgcn_wmma_f32_16x16x32_f16(false, a, false, b, (short)0, c, false, false);
}

// ---------------------------------------------------------------------------
// Weight convert + transpose: w [DIM][ncols] f32 -> wt [ncols][DIM] f16
// ---------------------------------------------------------------------------
__global__ void wcvt_transpose(const float* __restrict__ w, _Float16* __restrict__ wt,
                               int ncols) {
  int idx = blockIdx.x * 256 + threadIdx.x;
  int total = DIM * ncols;
  if (idx >= total) return;
  int k = idx / ncols;
  int j = idx - k * ncols;
  wt[(size_t)j * DIM + k] = (_Float16)w[idx];
}

// ---------------------------------------------------------------------------
// Q/K projection:  x[8192,512] f32 (cvt f16) @ Wt rows [cb..cb+63]
// wave = 16 rows x 64 cols; block = 8 waves (128 rows)
// grid (64, 16): blockIdx.y < 8 -> Q head y ; >= 8 -> K head y-8
// ---------------------------------------------------------------------------
__global__ void __launch_bounds__(256)
proj_qk(const float* __restrict__ x, const _Float16* __restrict__ Wt,
        _Float16* __restrict__ Qh, _Float16* __restrict__ Kh) {
  const int lane = threadIdx.x & 31;
  const int wave = threadIdx.x >> 5;
  const int q16  = lane & 15;
  const int hh   = lane >> 4;
  const int rb   = blockIdx.x * 128 + wave * 16;   // combined B*N row
  const int cb   = blockIdx.y * 64;                // qkv col base (0..1023)

  v8f acc[4];
#pragma unroll
  for (int t = 0; t < 4; ++t) acc[t] = zero8();

  const float* xr = x + (size_t)(rb + q16) * DIM;
#pragma unroll 4
  for (int k = 0; k < 16; ++k) {
    const float* xk = xr + k * 32;
    v16h a = cvt16(xk + 16 * hh);
#pragma unroll
    for (int t = 0; t < 4; ++t) {
      const _Float16* wr = Wt + (size_t)(cb + t * 16 + q16) * DIM + k * 32 + hh * 16;
      acc[t] = wmma16(a, ld16(wr), acc[t]);
    }
  }

  const int bb = rb >> 12;           // batch
  const int nb = rb & (SEQ - 1);     // n within batch
  const int hsel = blockIdx.y;
  _Float16* dst = (hsel < HEADS ? Qh : Kh) +
                  ((size_t)(bb * HEADS + (hsel & 7))) * SEQ * DIMH;
#pragma unroll
  for (int t = 0; t < 4; ++t)
#pragma unroll
    for (int r = 0; r < 8; ++r) {
      int n  = nb + r + 8 * hh;
      int dh = t * 16 + q16;
      dst[(size_t)n * DIMH + dh] = (_Float16)acc[t][r];
    }
}

// ---------------------------------------------------------------------------
// V projection, transposed:  D = Wv^T x x^T  -> Vt [B,H,64,SEQ] f16
// wave = 64 dh x 16 n; grid (64, 8): blockIdx.y = head
// ---------------------------------------------------------------------------
__global__ void __launch_bounds__(256)
proj_v(const float* __restrict__ x, const _Float16* __restrict__ Wt,
       _Float16* __restrict__ Vt) {
  const int lane = threadIdx.x & 31;
  const int wave = threadIdx.x >> 5;
  const int q16  = lane & 15;
  const int hh   = lane >> 4;
  const int rb   = blockIdx.x * 128 + wave * 16;   // combined B*N row (n dim)
  const int hv   = blockIdx.y;                     // head 0..7

  v8f acc[4];
#pragma unroll
  for (int d = 0; d < 4; ++d) acc[d] = zero8();

  const float* xr = x + (size_t)(rb + q16) * DIM;
#pragma unroll 4
  for (int k = 0; k < 16; ++k) {
    v16h bx = cvt2x8(xr + k * 32 + 8 * hh, xr + k * 32 + 16 + 8 * hh);   // B = x^T : lane = n col
#pragma unroll
    for (int d = 0; d < 4; ++d) {
      const _Float16* wr =
          Wt + (size_t)(1024 + hv * 64 + d * 16 + q16) * DIM + k * 32;
      acc[d] = wmma16(ld2x8(wr + 8 * hh, wr + 16 + 8 * hh), bx, acc[d]);
    }
  }

  const int bb = rb >> 12;
  const int nb = rb & (SEQ - 1);
  _Float16* dst = Vt + ((size_t)(bb * HEADS + hv)) * DIMH * SEQ;
#pragma unroll
  for (int d = 0; d < 4; ++d)
#pragma unroll
    for (int r = 0; r < 8; ++r) {
      int dh = d * 16 + r + 8 * hh;
      int n  = nb + q16;
      dst[(size_t)dh * SEQ + n] = (_Float16)acc[d][r];
    }
}

// ---------------------------------------------------------------------------
// Flash attention, LDS-staged + double-buffered.
// Block = 8 waves = 256 queries of one (b,h); wave = 32 queries (2 Q-tiles).
// Per 64-key tile: stage K[64x64] and V^T[64x64] through LDS (reg-pipelined,
// one barrier per tile).  S^T = K x Q^T (16 WMMAs), ONE softmax pass per
// 64-key tile (exp2 domain; packed-f16 P fragments built with packed
// cvt/shfl/select), out^T += V^T x P^T (16 WMMAs).
// grid (16, HEADS, 2), block 256.
// ---------------------------------------------------------------------------
__global__ void __launch_bounds__(256)
attn(const _Float16* __restrict__ Qh, const _Float16* __restrict__ Kh,
     const _Float16* __restrict__ Vt, _Float16* __restrict__ Oh) {
  __shared__ _Float16 Kls[2][64 * LPAD];
  __shared__ _Float16 Vls[2][64 * LPAD];

  const int lane = threadIdx.x & 31;
  const int wave = threadIdx.x >> 5;
  const int q16  = lane & 15;
  const int hh   = lane >> 4;
  const int b    = blockIdx.z;
  const int hd   = blockIdx.y;
  const int qb   = blockIdx.x * 256 + wave * 32;

  const size_t hoff = (size_t)(b * HEADS + hd) * SEQ * DIMH;
  const _Float16* Qp = Qh + hoff;       // [SEQ][64]
  const _Float16* Kp = Kh + hoff;       // [SEQ][64]
  const _Float16* Vp = Vt + hoff;       // [64][SEQ]

  // Copy mapping: wave w moves rows [8w, 8w+8); instr i covers rows 8w+i*4..+3,
  // lane -> row = 8w + i*4 + lane/8, col = (lane%8)*8 halves (16B chunk).
  const int crow = wave * 8 + (lane >> 3);
  const int ccol = (lane & 7) * 8;

  auto loadK = [&](int kb, int i) -> v8h {
    return *reinterpret_cast<const v8h*>(
        Kp + (size_t)(kb + crow + i * 4) * DIMH + ccol);
  };
  auto loadV = [&](int kb, int i) -> v8h {
    return *reinterpret_cast<const v8h*>(
        Vp + (size_t)(crow + i * 4) * SEQ + kb + ccol);
  };
  auto storeT = [&](int buf, int i, v8h kd, v8h vd) {
    *reinterpret_cast<v8h*>(&Kls[buf][(crow + i * 4) * LPAD + ccol]) = kd;
    *reinterpret_cast<v8h*>(&Vls[buf][(crow + i * 4) * LPAD + ccol]) = vd;
  };

  // Q^T fragments, pre-scaled by SCALE * log2(e) (softmax in base-2 domain).
  const _Float16 qs = (_Float16)(SCALE * 1.44269504f);
  v16h bq[QT][2];
#pragma unroll
  for (int qt = 0; qt < QT; ++qt)
#pragma unroll
    for (int c = 0; c < 2; ++c) {
      F16x16 u;
      u.v = ld16(Qp + (size_t)(qb + qt * 16 + q16) * DIMH + c * 32 + hh * 16);
#pragma unroll
      for (int e = 0; e < 16; ++e) u.e[e] = (_Float16)(u.e[e] * qs);
      bq[qt][c] = u.v;
    }

  v8f acc[QT][4];
#pragma unroll
  for (int qt = 0; qt < QT; ++qt)
#pragma unroll
    for (int d = 0; d < 4; ++d) acc[qt][d] = zero8();
  float mrow[QT] = {-1e30f, -1e30f};
  float lrow[QT] = {0.0f, 0.0f};

  // Prologue: tile0 -> LDS buf0; tile1 -> staged regs.
  {
    v8h k0 = loadK(0, 0), k1 = loadK(0, 1);
    v8h v0 = loadV(0, 0), v1 = loadV(0, 1);
    storeT(0, 0, k0, v0);
    storeT(0, 1, k1, v1);
  }
  v8h sK0 = loadK(64, 0), sK1 = loadK(64, 1);
  v8h sV0 = loadV(64, 0), sV1 = loadV(64, 1);
  __syncthreads();

  for (int it = 0; it < KT64; ++it) {
    const int buf = it & 1;

    // Issue global loads for tile it+2 (fresh regs, consumed next iter).
    v8h nK0, nK1, nV0, nV1;
    const bool pf = (it + 2 < KT64);
    if (pf) {
      const int kb2 = (it + 2) * 64;
      nK0 = loadK(kb2, 0); nK1 = loadK(kb2, 1);
      nV0 = loadV(kb2, 0); nV1 = loadV(kb2, 1);
    }
    // Store tile it+1 (held in regs) into the idle buffer; its previous
    // readers finished at the barrier that ended iteration it-1.
    if (it + 1 < KT64) {
      storeT(buf ^ 1, 0, sK0, sV0);
      storeT(buf ^ 1, 1, sK1, sV1);
    }

    // ---- S^T phase: 4 key subtiles x 2 dh chunks x QT -> 16 WMMAs ----
    v8f sS[QT][4];
#pragma unroll
    for (int qt = 0; qt < QT; ++qt)
#pragma unroll
      for (int t = 0; t < 4; ++t) sS[qt][t] = zero8();

#pragma unroll
    for (int c = 0; c < 2; ++c)
#pragma unroll
      for (int t = 0; t < 4; ++t) {
        const _Float16* krow = &Kls[buf][(t * 16 + q16) * LPAD + c * 32];
        v16h ak = ld16(krow + 16 * hh);
#pragma unroll
        for (int qt = 0; qt < QT; ++qt)
          sS[qt][t] = wmma16(ak, bq[qt][c], sS[qt][t]);
      }

    // ---- one softmax pass over all 64 keys ----
    F16x16 bp[QT][2];
    float alpha[QT];
#pragma unroll
    for (int qt = 0; qt < QT; ++qt) {
      float lmax = -1e30f;
#pragma unroll
      for (int t = 0; t < 4; ++t)
#pragma unroll
        for (int r = 0; r < 8; ++r) lmax = fmaxf(lmax, sS[qt][t][r]);
      lmax = fmaxf(lmax, __shfl_xor(lmax, 16, 32));
      const float mnew = fmaxf(mrow[qt], lmax);
      alpha[qt] = exp2f(mrow[qt] - mnew);

      float lsum = 0.0f;
#pragma unroll
      for (int c2 = 0; c2 < 2; ++c2) {
        float pa[8], pb[8];
#pragma unroll
        for (int r = 0; r < 8; ++r) {
          pa[r] = exp2f(sS[qt][2 * c2][r] - mnew);
          pb[r] = exp2f(sS[qt][2 * c2 + 1][r] - mnew);
          lsum += pa[r] + pb[r];
        }
        // packed P^T fragment: u[j] (j<4) = own subtile-even / partner subtile-odd halves (hh=0/1),
        // u[4+j] = partner subtile-even / own subtile-odd halves (see layout derivation).
        unsigned pka[4], pkb[4];
#pragma unroll
        for (int j = 0; j < 4; ++j) {
          pka[j] = pack2(pa[2 * j], pa[2 * j + 1]);
          pkb[j] = pack2(pb[2 * j], pb[2 * j + 1]);
        }
#pragma unroll
        for (int j = 0; j < 4; ++j) {
          unsigned xa = (unsigned)__shfl_xor((int)pka[j], 16, 32);
          unsigned xb = (unsigned)__shfl_xor((int)pkb[j], 16, 32);
          bp[qt][c2].u[j]     = hh ? xb : pka[j];
          bp[qt][c2].u[4 + j] = hh ? pkb[j] : xa;
        }
      }
      lsum += __shfl_xor(lsum, 16, 32);
      lrow[qt] = alpha[qt] * lrow[qt] + lsum;
      mrow[qt] = mnew;
    }

    // rescale accumulators once per 64-key tile
#pragma unroll
    for (int qt = 0; qt < QT; ++qt) {
      v8f am;
#pragma unroll
      for (int i = 0; i < 8; ++i) am[i] = alpha[qt];
#pragma unroll
      for (int d = 0; d < 4; ++d) acc[qt][d] = acc[qt][d] * am;
    }

    // ---- PV phase: out^T += V^T x P^T (2 key chunks x 4 dh tiles x QT) ----
#pragma unroll
    for (int c2 = 0; c2 < 2; ++c2)
#pragma unroll
      for (int d = 0; d < 4; ++d) {
        const _Float16* vrow = &Vls[buf][(d * 16 + q16) * LPAD + c2 * 32];
        v16h av = ld16(vrow + 16 * hh);
#pragma unroll
        for (int qt = 0; qt < QT; ++qt)
          acc[qt][d] = wmma16(av, bp[qt][c2].v, acc[qt][d]);
      }

    if (pf) { sK0 = nK0; sK1 = nK1; sV0 = nV0; sV1 = nV1; }
    __syncthreads();
  }

#pragma unroll
  for (int qt = 0; qt < QT; ++qt) {
    const float inv = 1.0f / lrow[qt];
    _Float16* orow =
        Oh + (size_t)(b * SEQ + qb + qt * 16 + q16) * DIM + hd * DIMH;
#pragma unroll
    for (int d = 0; d < 4; ++d)
#pragma unroll
      for (int r = 0; r < 8; ++r)
        orow[d * 16 + r + 8 * hh] = (_Float16)(acc[qt][d][r] * inv);
  }
}

// ---------------------------------------------------------------------------
// Output projection: out = Oh @ w_out + b_out  (f32 result)
// wave = 16 rows x 64 cols; grid (64, 8)
// ---------------------------------------------------------------------------
__global__ void __launch_bounds__(256)
proj_out(const _Float16* __restrict__ Oh, const _Float16* __restrict__ Wt,
         const float* __restrict__ bias, float* __restrict__ out) {
  const int lane = threadIdx.x & 31;
  const int wave = threadIdx.x >> 5;
  const int q16  = lane & 15;
  const int hh   = lane >> 4;
  const int rb   = blockIdx.x * 128 + wave * 16;
  const int cb   = blockIdx.y * 64;

  v8f acc[4];
#pragma unroll
  for (int t = 0; t < 4; ++t) acc[t] = zero8();

  const _Float16* orow = Oh + (size_t)(rb + q16) * DIM;
#pragma unroll 4
  for (int k = 0; k < 16; ++k) {
    v16h a = ld16(orow + k * 32 + 16 * hh);
#pragma unroll
    for (int t = 0; t < 4; ++t) {
      const _Float16* wr = Wt + (size_t)(cb + t * 16 + q16) * DIM + k * 32 + hh * 16;
      acc[t] = wmma16(a, ld16(wr), acc[t]);
    }
  }

#pragma unroll
  for (int t = 0; t < 4; ++t)
#pragma unroll
    for (int r = 0; r < 8; ++r) {
      int row = rb + r + 8 * hh;
      int j   = cb + t * 16 + q16;
      out[(size_t)row * DIM + j] = acc[t][r] + bias[j];
    }
}

// ---------------------------------------------------------------------------
// Host launcher. Workspace layout (f16 elements):
//   Wqkv_t [1536*512] | Wout_t [512*512] | Qh | Kh | Vt | Oh  (each 4,194,304)
// Total ~35.7 MB.
// ---------------------------------------------------------------------------
extern "C" void kernel_launch(void* const* d_in, const int* in_sizes, int n_in,
                              void* d_out, int out_size, void* d_ws, size_t ws_size,
                              hipStream_t stream) {
  (void)in_sizes; (void)n_in; (void)out_size; (void)ws_size;
  const float* x     = (const float*)d_in[0];   // [2,4096,512]
  const float* w_qkv = (const float*)d_in[1];   // [512,1536]
  const float* w_out = (const float*)d_in[2];   // [512,512]
  const float* b_out = (const float*)d_in[3];   // [512]
  float* out = (float*)d_out;

  _Float16* ws   = (_Float16*)d_ws;
  _Float16* WqkT = ws;                          // 786432
  _Float16* WoT  = WqkT + (size_t)1536 * 512;   // 262144
  _Float16* Qh   = WoT + (size_t)512 * 512;     // 4194304 each
  _Float16* Kh   = Qh + (size_t)2 * HEADS * SEQ * DIMH;
  _Float16* Vt   = Kh + (size_t)2 * HEADS * SEQ * DIMH;
  _Float16* Oh   = Vt + (size_t)2 * HEADS * SEQ * DIMH;

  wcvt_transpose<<<(512 * 1536 + 255) / 256, 256, 0, stream>>>(w_qkv, WqkT, 1536);
  wcvt_transpose<<<(512 * 512 + 255) / 256, 256, 0, stream>>>(w_out, WoT, 512);

  proj_qk<<<dim3(64, 16), 256, 0, stream>>>(x, WqkT, Qh, Kh);
  proj_v <<<dim3(64, 8),  256, 0, stream>>>(x, WqkT, Vt);

  attn<<<dim3(SEQ / 256, HEADS, 2), 256, 0, stream>>>(Qh, Kh, Vt, Oh);

  proj_out<<<dim3(64, 8), 256, 0, stream>>>(Oh, WoT, b_out, out);
}
